// LSTMCell_11759620456639
// MI455X (gfx1250) — matrix operand order repaired and checked by reading, not compiled, not run on hardware
//
#include <hip/hip_runtime.h>
#include <hip/hip_bf16.h>

// LSTM cell, M=16384 (batch), D_S=D_IN=1024.
// Fused GEMM: X=[input|hidden] (16384x2048) x W^T (4096x2048), gate-interleaved
// N (n = s*4+g) so each 128x128 C tile holds all 4 gates for 32 s-columns ->
// LSTM epilogue fused into the GEMM via an LDS transpose. ~275 GFLOP vs
// ~0.25GB HBM traffic => compute bound => v_wmma_f32_16x16x32_bf16 datapath,
// with GLOBAL_LOAD_ASYNC_TO_LDS_B128 double-buffered staging (ASYNCcnt).

#define BATCH 16384
#define DS    1024
#define KTOT  2048   // D_IN + D_S
#define NTOT  4096   // 4 gates * D_S, interleaved n = s*4 + g

#define BM 128
#define BN 128
#define BK 64
#define LDSROW 72                 // ushorts per staged row: 64 data + 8 pad
#define CROW   (BN + 4)           // f32 tile row stride (132 floats, 16B mult)

// LDS arena: two stage buffers (A:18KB + B:18KB each), epilogue C tile overlaps.
#define STAGE     ((BM + BN) * LDSROW * 2)   // 36864 per buffer
#define BOFF      (BM * LDSROW * 2)          // B offset inside a buffer: 18432
#define LDS_BYTES (2 * STAGE)                // 73728 >= C tile (67584)

typedef __attribute__((ext_vector_type(16))) __bf16        bf16x16;
typedef __attribute__((ext_vector_type(8)))  float         f32x8;
typedef __attribute__((ext_vector_type(4)))  float         f32x4;
typedef __attribute__((ext_vector_type(4)))  unsigned int  u32x4;
typedef __attribute__((ext_vector_type(2)))  unsigned int  u32x2;

union FragU {
  bf16x16 v;
  u32x4   q[2];
};

__device__ __forceinline__ unsigned int f2bf(float f) {
  unsigned int u = __float_as_uint(f);
  u += 0x7FFFu + ((u >> 16) & 1u);    // round-to-nearest-even
  return u >> 16;
}

__device__ __forceinline__ float sigf(float x) {
  return 1.0f / (1.0f + __expf(-x));
}

// ---- Kernel 1: pack [input | hidden] -> bf16 Xc [BATCH x KTOT] -------------
__global__ void cvt_x_kernel(const float* __restrict__ x,
                             const float* __restrict__ h,
                             unsigned short* __restrict__ Xc) {
  const int idx4 = blockIdx.x * blockDim.x + threadIdx.x;  // 4 elems/thread
  const int k = (idx4 << 2) & (KTOT - 1);
  const int b = (idx4 << 2) >> 11;                         // /KTOT
  const float* src = (k < DS) ? (x + (size_t)b * DS + k)
                              : (h + (size_t)b * DS + (k - DS));
  f32x4 v = *(const f32x4*)src;
  u32x2 o;
  o[0] = f2bf(v[0]) | (f2bf(v[1]) << 16);
  o[1] = f2bf(v[2]) | (f2bf(v[3]) << 16);
  *(u32x2*)(Xc + ((size_t)idx4 << 2)) = o;
}

// ---- Kernel 2: pack [Wx | Wh] -> bf16 Wc [NTOT x KTOT], n = s*4 + g -------
__global__ void cvt_w_kernel(const float* __restrict__ Wx,
                             const float* __restrict__ Wh,
                             unsigned short* __restrict__ Wc) {
  const int idx4 = blockIdx.x * blockDim.x + threadIdx.x;
  const int k = (idx4 << 2) & (KTOT - 1);
  const int n = (idx4 << 2) >> 11;
  const int g = n & 3;          // gate interleaved in low bits of N
  const int s = n >> 2;
  const float* src = (k < DS)
      ? (Wx + ((size_t)g * DS + s) * DS + k)
      : (Wh + ((size_t)g * DS + s) * DS + (k - DS));
  f32x4 v = *(const f32x4*)src;
  u32x2 o;
  o[0] = f2bf(v[0]) | (f2bf(v[1]) << 16);
  o[1] = f2bf(v[2]) | (f2bf(v[3]) << 16);
  *(u32x2*)(Wc + ((size_t)idx4 << 2)) = o;
}

// ---- Kernel 3: WMMA GEMM + fused LSTM epilogue ----------------------------
// 256 threads = 8 wave32, waves 4(M) x 2(N); each wave: 32x64 C tile
// = 2x4 accumulators of v_wmma_f32_16x16x32_bf16. Staging is double-buffered
// GLOBAL_LOAD_ASYNC_TO_LDS_B128 (no VGPR round-trip, ASYNCcnt tracked).
__global__ __launch_bounds__(256)
void gemm_lstm_kernel(const unsigned short* __restrict__ Xc,
                      const unsigned short* __restrict__ Wc,
                      const float* __restrict__ cell,
                      const float* __restrict__ bx,
                      const float* __restrict__ bh,
                      float* __restrict__ out) {
  extern __shared__ __align__(16) char smem[];

  const int t    = threadIdx.x;
  const int lane = t & 31;
  const int wave = t >> 5;
  const int wm   = wave & 3;    // M sub-tile: wm*32
  const int wn   = wave >> 2;   // N sub-tile: wn*64
  const int ln   = lane & 15;
  const int hi   = lane >> 4;   // half-wave select for WMMA K layout

  const int mBase = blockIdx.y * BM;
  const int nBase = blockIdx.x * BN;

  // Staging: thread t owns 32 ushorts (64B = 4 x b128) of one row of A and B.
  const int rowS = t >> 1;              // 0..127
  const int kcol = (t & 1) * 32;        // ushort col within BK=64 row
  const unsigned short* gA = Xc + (size_t)(mBase + rowS) * KTOT + kcol;
  const unsigned short* gB = Wc + (size_t)(nBase + rowS) * KTOT + kcol;
  const unsigned int stOff = (unsigned int)(rowS * (LDSROW * 2) + kcol * 2);

  f32x8 acc[2][4] = {};

  // Issue one K-tile of async global->LDS copies into buffer bufSel.
  auto issue_stage = [&](int kt, int bufSel) {
    const unsigned int aOff = (unsigned int)(bufSel * STAGE) + stOff;
    const unsigned int bOff = aOff + (unsigned int)BOFF;
    const unsigned short* pA = gA + (size_t)kt * BK;
    const unsigned short* pB = gB + (size_t)kt * BK;
#pragma unroll
    for (int c = 0; c < 4; ++c) {
      asm volatile("global_load_async_to_lds_b128 %0, %1, off"
                   :: "v"(aOff + c * 16), "v"(pA + c * 8) : "memory");
      asm volatile("global_load_async_to_lds_b128 %0, %1, off"
                   :: "v"(bOff + c * 16), "v"(pB + c * 8) : "memory");
    }
  };

  const int NK = KTOT / BK;             // 32 iterations
  issue_stage(0, 0);
  asm volatile("s_wait_asynccnt 0x0" ::: "memory");
  __syncthreads();

  for (int kt = 0; kt < NK; ++kt) {
    const int buf = kt & 1;
    if (kt + 1 < NK) {
      issue_stage(kt + 1, buf ^ 1);     // fill other buffer asynchronously
      if (kt + 2 < NK) {                // L2 prefetch (global_prefetch_b8)
        __builtin_prefetch(gA + (size_t)(kt + 2) * BK, 0, 1);
        __builtin_prefetch(gB + (size_t)(kt + 2) * BK, 0, 1);
      }
    }

    auto As = (const unsigned short (*)[LDSROW])(smem + buf * STAGE);
    auto Bs = (const unsigned short (*)[LDSROW])(smem + buf * STAGE + BOFF);

#pragma unroll
    for (int ks = 0; ks < BK; ks += 32) {   // two WMMA K-steps per stage
      FragU fa[2], fb[4];
#pragma unroll
      for (int i = 0; i < 2; ++i) {
        // A 16x32: lanes 0-15 (row M=ln) K {0..7,16..23}; lanes 16-31
        // K {8..15,24..31} -> two b128 LDS loads per lane.
        const unsigned short* pa = &As[wm * 32 + i * 16 + ln][ks + hi * 8];
        fa[i].q[0] = *(const u32x4*)(pa);
        fa[i].q[1] = *(const u32x4*)(pa + 16);
      }
#pragma unroll
      for (int j = 0; j < 4; ++j) {
        // B 32x16: lane col N=ln, K {hi*8..hi*8+7, 16+hi*8..16+hi*8+7}.
        const unsigned short* pb = &Bs[wn * 64 + j * 16 + ln][ks + hi * 8];
        fb[j].q[0] = *(const u32x4*)(pb);
        fb[j].q[1] = *(const u32x4*)(pb + 16);
      }
#pragma unroll
      for (int j = 0; j < 4; ++j)
#pragma unroll
        for (int i = 0; i < 2; ++i)
          acc[i][j] = __builtin_amdgcn_wmma_f32_16x16x32_bf16(
              false, fa[i].v, false, fb[j].v, (short)0, acc[i][j], false, false);
    }

    if (kt + 1 < NK)
      asm volatile("s_wait_asynccnt 0x0" ::: "memory");  // next buffer landed
    __syncthreads();
  }

  // ---- Fused epilogue ----
  // Accumulator layout: VGPR e holds (M = e + hi*8, N = ln). Dump the 128x128
  // f32 tile to LDS (overlapping the dead stage buffers), then each thread
  // reads 4 consecutive gate values per (row, s) with one b128 LDS load.
  auto Cs = (float (*)[CROW])(smem);
#pragma unroll
  for (int i = 0; i < 2; ++i)
#pragma unroll
    for (int j = 0; j < 4; ++j) {
      const int col = wn * 64 + j * 16 + ln;
#pragma unroll
      for (int e = 0; e < 8; ++e)
        Cs[wm * 32 + i * 16 + hi * 8 + e][col] = acc[i][j][e];
    }
  __syncthreads();

  const int sc    = t & 31;             // s-column within tile (0..31)
  const int sglob = (nBase >> 2) + sc;  // global s index (0..1023)
  const int wv    = t >> 5;

  f32x4 BX, BH;                         // biases, gate-major [4,1024]
#pragma unroll
  for (int g = 0; g < 4; ++g) {
    BX[g] = bx[g * DS + sglob];
    BH[g] = bh[g * DS + sglob];
  }

#pragma unroll
  for (int it = 0; it < 16; ++it) {
    const int r = it * 8 + wv;          // 0..127; lanes -> consecutive s
    const size_t gidx = (size_t)(mBase + r) * DS + sglob;
    f32x4 P = *(const f32x4*)&Cs[r][sc * 4];   // gates f,i,c,o
    float pf = P[0] + BX[0] + BH[0];
    float pi = P[1] + BX[1] + BH[1];
    float pc = P[2] + BX[2] + BH[2];
    float po = P[3] + BX[3] + BH[3];
    float f  = sigf(pf);
    float ig = sigf(pi) * tanhf(pc);
    float cn = cell[gidx] * f + ig;
    out[gidx] = cn;                                    // new_cell_state
    out[(size_t)BATCH * DS + gidx] = tanhf(cn) * sigf(po);  // output
  }
}

extern "C" void kernel_launch(void* const* d_in, const int* in_sizes, int n_in,
                              void* d_out, int out_size, void* d_ws, size_t ws_size,
                              hipStream_t stream) {
  const float* x    = (const float*)d_in[0];  // input       [B, 1024]
  const float* cell = (const float*)d_in[1];  // cell_state  [B, 1024]
  const float* h    = (const float*)d_in[2];  // hidden      [B, 1024]
  const float* Wx   = (const float*)d_in[3];  // [4,1024,1024]
  const float* bx   = (const float*)d_in[4];  // [4,1024]
  const float* Wh   = (const float*)d_in[5];  // [4,1024,1024]
  const float* bh   = (const float*)d_in[6];  // [4,1024]
  float* out = (float*)d_out;                 // [new_cell | output]

  // Workspace: Xc bf16 (64MB) | Wc bf16 (16MB)
  char* wsb = (char*)d_ws;
  const size_t xcBytes = (size_t)BATCH * KTOT * 2;
  unsigned short* Xc = (unsigned short*)wsb;
  unsigned short* Wc = (unsigned short*)(wsb + xcBytes);

  cvt_x_kernel<<<(BATCH * (KTOT / 4)) / 256, 256, 0, stream>>>(x, h, Xc);
  cvt_w_kernel<<<(NTOT  * (KTOT / 4)) / 256, 256, 0, stream>>>(Wx, Wh, Wc);

  dim3 grid(NTOT / BN, BATCH / BM);   // (32, 128)
  gemm_lstm_kernel<<<grid, 256, LDS_BYTES, stream>>>(Xc, Wc, cell, bx, bh, out);
}
